// GCU_24249385353356
// MI455X (gfx1250) — matrix operand order repaired and checked by reading, not compiled, not run on hardware
//
#include <hip/hip_runtime.h>

typedef __attribute__((ext_vector_type(16))) _Float16 v16h;
typedef __attribute__((ext_vector_type(8)))  float    v8f;

union H16 { v16h v; _Float16 h[16]; };
union Q2H { uint4 q[2]; v16h v; };

#define VSTRIDE 196            // padded floats per node row (196 % 64 = 4 -> bank spread)
#define ZSTRIDE 68             // padded floats per z_ row (272 B: 16B-aligned, bank-spread)
#define FRAG_LANE_BYTES 48     // 32B payload + 16B pad: keeps 16B align, spreads banks
#define FRAG_BYTES (32 * FRAG_LANE_BYTES)
#define NFRAG 40               // Bn:0..15 (c*4+t)  Bc:16..23 (16+c*4+t)  Bo:24..39 (24+c*4+t)

#define WMMA(A, B, C) \
  __builtin_amdgcn_wmma_f32_16x16x32_f16(false, (A), false, (B), (short)0, (C), false, false)

__device__ __forceinline__ float celu1(float x) {
  return x > 0.0f ? x : (__expf(x) - 1.0f);
}

// Gather one B fragment (32x16 f16) from a row-major f32 [K x 64] weight matrix.
// kbase = k0 + 16*half; lanes 0-15: K = {k0..k0+7, k0+16..k0+23}, lanes 16-31: +8; col = lane%16.
__device__ __forceinline__ v16h bfrag_gather(const float* __restrict__ W, int kbase, int col) {
  H16 r;
#pragma unroll
  for (int e = 0; e < 16; ++e) r.h[e] = (_Float16)W[(kbase - 8 * ((kbase >> 4) & 1) + (e < 8 ? e : e + 8)) * 64 + col];
  return r.v;
}

__device__ __forceinline__ void store_frag(char* base, int f, int lane, v16h v) {
  Q2H u; u.v = v;
  uint4* p = (uint4*)(base + f * FRAG_BYTES + lane * FRAG_LANE_BYTES);
  p[0] = u.q[0];
  p[1] = u.q[1];
}

__device__ __forceinline__ v16h load_frag(const char* base, int f, int lane) {
  const uint4* p = (const uint4*)(base + f * FRAG_BYTES + lane * FRAG_LANE_BYTES);
  Q2H u;
  u.q[0] = p[0];
  u.q[1] = p[1];
  return u.v;
}

// A fragment (16x32 f16) from a 16B-aligned f32 row; c0 is a multiple of 8 floats.
// lane<16 -> K = {c0..c0+7, c0+16..c0+23}; lane>=16 shifted by +8 (caller folds into c0).
__device__ __forceinline__ v16h afrag_f32(const float* rp, int c0, bool valid) {
  float4 q0 = {0.f, 0.f, 0.f, 0.f}, q1 = q0, q2 = q0, q3 = q0;
  if (valid) {
    const float4* p0 = (const float4*)(rp + c0);
    const float4* p1 = (const float4*)(rp + c0 + 16);
    q0 = p0[0]; q1 = p0[1];
    q2 = p1[0]; q3 = p1[1];
  }
  H16 a;
  a.h[0]  = (_Float16)q0.x;  a.h[1]  = (_Float16)q0.y;
  a.h[2]  = (_Float16)q0.z;  a.h[3]  = (_Float16)q0.w;
  a.h[4]  = (_Float16)q1.x;  a.h[5]  = (_Float16)q1.y;
  a.h[6]  = (_Float16)q1.z;  a.h[7]  = (_Float16)q1.w;
  a.h[8]  = (_Float16)q2.x;  a.h[9]  = (_Float16)q2.y;
  a.h[10] = (_Float16)q2.z;  a.h[11] = (_Float16)q2.w;
  a.h[12] = (_Float16)q3.x;  a.h[13] = (_Float16)q3.y;
  a.h[14] = (_Float16)q3.z;  a.h[15] = (_Float16)q3.w;
  return a.v;
}

__launch_bounds__(256, 1)
__global__ void gcu_wmma_kernel(const float* __restrict__ z,
                                const float* __restrict__ z_,
                                const float* __restrict__ W_cur,
                                const float* __restrict__ b_cur,
                                const float* __restrict__ W_nbr,
                                const float* __restrict__ b_nbr,
                                const float* __restrict__ W_out,
                                const float* __restrict__ b_out,
                                float* __restrict__ out,
                                int nNodes)
{
  extern __shared__ float smem[];
  float (*sB)[64]         = (float (*)[64])smem;                         // [3][64] biases
  float (*sVcat)[VSTRIDE] = (float (*)[VSTRIDE])(smem + 192);            // [8] rows: v | v_ | z
  float* sZ               = smem + 192 + 8 * VSTRIDE;                    // [8][32*ZSTRIDE]
  char*  sF               = (char*)(smem + 192 + 8 * VSTRIDE + 8 * 32 * ZSTRIDE);

  const int tid   = threadIdx.x;
  const int lane  = tid & 31;
  const int wslot = tid >> 5;
  const int half  = lane >> 4;
  const int l16   = lane & 15;
  const int t4    = wslot & 3;

  if (tid < 64)       sB[0][tid]       = b_cur[tid];
  else if (tid < 128) sB[1][tid - 64]  = b_nbr[tid - 64];
  else if (tid < 192) sB[2][tid - 128] = b_out[tid - 128];

  // ---- one shared, pre-swizzled copy of all weight B-fragments (wave 0 builds it) ----
  if (wslot == 0) {
#pragma unroll
    for (int c = 0; c < 4; ++c)
#pragma unroll
      for (int t = 0; t < 4; ++t)
        store_frag(sF, c * 4 + t, lane, bfrag_gather(W_nbr, 32 * c + 16 * half, 16 * t + l16));
#pragma unroll
    for (int c = 0; c < 2; ++c)
#pragma unroll
      for (int t = 0; t < 4; ++t)
        store_frag(sF, 16 + c * 4 + t, lane, bfrag_gather(W_cur, 32 * c + 16 * half, 16 * t + l16));
#pragma unroll
    for (int c = 0; c < 4; ++c)
#pragma unroll
      for (int t = 0; t < 4; ++t)
        store_frag(sF, 24 + c * 4 + t, lane, bfrag_gather(W_out, 32 * c + 16 * half, 16 * t + l16));
  }
  __syncthreads();

  const int groupStride = gridDim.x * 8;
  for (int base = blockIdx.x * 8; base < nNodes; base += groupStride) {
    const int node = base + wslot;                    // wave-uniform
    if (node < nNodes) {
      const float* zrow = z + (size_t)node * 64;
      float* sZw = sZ + wslot * (32 * ZSTRIDE);

      // -- async-copy this node's z_ tile (32x64 f32, 8 KB) into padded LDS --
      {
        const unsigned lrow = lane >> 4;              // 2 rows per issue
        const unsigned lcol = (lane & 15) * 16;
        unsigned ldsAddr = (unsigned)(uintptr_t)sZw + lrow * (ZSTRIDE * 4) + lcol;
        unsigned long long gAddr =
            (unsigned long long)(uintptr_t)(z_ + (size_t)node * 2048) + lrow * 256 + lcol;
#pragma unroll
        for (int i = 0; i < 16; ++i) {
          asm volatile("global_load_async_to_lds_b128 %0, %1, off"
                       :: "v"(ldsAddr), "v"(gAddr) : "memory");
          ldsAddr += 2 * ZSTRIDE * 4;
          gAddr   += 512;
        }
      }
      // stage z for the stage-2 GEMM
      sVcat[wslot][128 + lane * 2 + 0] = zrow[lane * 2 + 0];
      sVcat[wslot][128 + lane * 2 + 1] = zrow[lane * 2 + 1];
      // prefetch next iteration's z_ toward the caches
      if (node + groupStride < nNodes) {
        const float* pf = z_ + (size_t)(node + groupStride) * 2048 + (size_t)lane * 64;
        __builtin_prefetch(pf, 0, 0);
        __builtin_prefetch(pf + 1024, 0, 0);
      }

      v8f acc[2][4] = {};
      // K-chunks 0-1: broadcast z part (A identical for both M-tiles -> CSE to 8 WMMAs)
#pragma unroll
      for (int c = 0; c < 2; ++c) {
        v16h a = afrag_f32(zrow, 32 * c + 8 * half, true);
#pragma unroll
        for (int t = 0; t < 4; ++t) {
          v16h b = load_frag(sF, c * 4 + t, lane);
          acc[0][t] = WMMA(a, b, acc[0][t]);
          acc[1][t] = WMMA(a, b, acc[1][t]);
        }
      }

      asm volatile("s_wait_asynccnt 0" ::: "memory");  // z_ tile landed in LDS

      // K-chunks 2-3: neighbor features from LDS
#pragma unroll
      for (int c = 0; c < 2; ++c) {
        const int c0 = 32 * c + 8 * half;
        const float* r0 = sZw + (size_t)l16 * ZSTRIDE;   // rows 0..15
        const float* r1 = r0 + 16 * ZSTRIDE;             // rows 16..31
        v16h a0 = afrag_f32(r0, c0, true);
        v16h a1 = afrag_f32(r1, c0, true);
#pragma unroll
        for (int t = 0; t < 4; ++t) {
          v16h b = load_frag(sF, (c + 2) * 4 + t, lane);
          acc[0][t] = WMMA(a0, b, acc[0][t]);
          acc[1][t] = WMMA(a1, b, acc[1][t]);
        }
      }

      // bias + CELU + neighbor-sum -> v_
#pragma unroll
      for (int t = 0; t < 4; ++t) {
        const float bn = sB[1][16 * t + l16];
        float s = 0.0f;
#pragma unroll
        for (int j = 0; j < 8; ++j) {
          s += celu1(acc[0][t][j] + bn);
          s += celu1(acc[1][t][j] + bn);
        }
        s += __shfl_xor(s, 16, 32);
        if (lane < 16) sVcat[wslot][64 + 16 * t + lane] = s;
      }
    }
    __syncthreads();

    // ---- stage 2: v = CELU(Z[8x64] @ W_cur + b_cur) via WMMA (waves 0-3, one N-tile each) ----
    if (wslot < 4) {
      v8f P = {};
#pragma unroll
      for (int c = 0; c < 2; ++c) {
        v16h a = afrag_f32(&sVcat[l16 & 7][128], 32 * c + 8 * half, l16 < 8);
        v16h b = load_frag(sF, 16 + c * 4 + t4, lane);
        P = WMMA(a, b, P);
      }
      if (lane < 16) {                  // rows 0-7 (the 8 nodes) live in lanes 0-15
        const float bc = sB[0][16 * t4 + lane];
#pragma unroll
        for (int j = 0; j < 8; ++j)
          sVcat[j][16 * t4 + lane] = celu1(P[j] + bc);
      }
    }
    __syncthreads();

    // ---- stage 3: out = Vcat[8x128] @ W_out + b_out via WMMA (waves 0-3) ----
    if (wslot < 4) {
      v8f O = {};
#pragma unroll
      for (int c = 0; c < 4; ++c) {
        v16h a = afrag_f32(&sVcat[l16 & 7][0], 32 * c + 8 * half, l16 < 8);
        v16h b = load_frag(sF, 24 + c * 4 + t4, lane);
        O = WMMA(a, b, O);
      }
      if (lane < 16) {
        const float bo = sB[2][16 * t4 + lane];
#pragma unroll
        for (int j = 0; j < 8; ++j)
          if (base + j < nNodes)
            out[(size_t)(base + j) * 64 + 16 * t4 + lane] = O[j] + bo;
      }
    }
    __syncthreads();    // protect sVcat/sZ reuse next iteration
  }
}

extern "C" void kernel_launch(void* const* d_in, const int* in_sizes, int n_in,
                              void* d_out, int out_size, void* d_ws, size_t ws_size,
                              hipStream_t stream) {
  const float* z     = (const float*)d_in[0];
  const float* z_n   = (const float*)d_in[1];
  const float* W_cur = (const float*)d_in[2];
  const float* b_cur = (const float*)d_in[3];
  const float* W_nbr = (const float*)d_in[4];
  const float* b_nbr = (const float*)d_in[5];
  const float* W_out = (const float*)d_in[6];
  const float* b_out = (const float*)d_in[7];
  float* out = (float*)d_out;

  const int nNodes = in_sizes[0] / 64;
  int blocks = (nNodes + 7) / 8;
  if (blocks > 768) blocks = 768;
  const size_t smemBytes =
      (size_t)(192 + 8 * VSTRIDE + 8 * 32 * ZSTRIDE) * sizeof(float) +
      (size_t)NFRAG * FRAG_BYTES;
  gcu_wmma_kernel<<<blocks, 256, smemBytes, stream>>>(z, z_n, W_cur, b_cur, W_nbr, b_nbr,
                                                      W_out, b_out, out, nNodes);
}
